// LaMemoRelLearnableMultiHeadAttn_8220567404626
// MI455X (gfx1250) — matrix operand order repaired and checked by reading, not compiled, not run on hardware
//
#include <hip/hip_runtime.h>
#include <stdint.h>

// LaMemo relative multi-head attention for MI455X (gfx1250, wave32).
// All GEMMs via v_wmma_f32_16x16x32_bf16 (bf16 in, f32 accumulate).
// GEMM A-tiles stream through LDS with global_load_async_to_lds_b128
// (ASYNCcnt) and double buffering; weights are pre-transposed so B
// fragments come straight from L2 with two 16B reads per lane.

typedef __attribute__((ext_vector_type(16))) __bf16 v16bf;
typedef __attribute__((ext_vector_type(8)))  float  v8f;

#define DEV __device__ __forceinline__

struct U4x2 { uint4 lo, hi; };

DEV v16bf frag_cast(uint4 lo, uint4 hi) {
  U4x2 t{lo, hi};
  return __builtin_bit_cast(v16bf, t);
}

DEV uint16_t f2bf(float f) {  // round-to-nearest-even f32 -> bf16
  uint32_t u = __float_as_uint(f);
  uint32_t r = (u + 0x7FFFu + ((u >> 16) & 1u)) >> 16;
  return (uint16_t)r;
}

// Async copy of 16 bytes from global to LDS (per active lane).
// VDST carries the per-lane LDS byte address (ISA: async load fields).
DEV void async_load_b128(uint32_t lds_byte_off, const void* gaddr) {
  asm volatile("global_load_async_to_lds_b128 %0, %1, off"
               :: "v"(lds_byte_off), "v"(gaddr) : "memory");
}
DEV void wait_async0() { asm volatile("s_wait_asynccnt 0x0" ::: "memory"); }

// A fragment: 16(M) x 32(K) bf16. Lane l holds row (l&15); element e holds
// k = (e>>3)*16 + (l>>4)*8 + (e&7)   (ISA 7.12.2 16-bit A layout).
DEV v16bf load_a_frag(const uint16_t* base, int ldk) {
  int lane = threadIdx.x & 31;
  const uint16_t* p = base + (size_t)(lane & 15) * ldk + ((lane >> 4) << 3);
  uint4 lo = *(const uint4*)(p);
  uint4 hi = *(const uint4*)(p + 16);
  return frag_cast(lo, hi);
}

// B fragment: 32(K) x 16(N) bf16 sourced from an N-major buffer (row n
// contiguous in k). Lane l holds col (l&15); element e holds
// k = (e>>3)*16 + (l>>4)*8 + (e&7)   (matches A layout) -> two 16B reads.
DEV v16bf load_bt_frag(const uint16_t* base, int ldk) {
  int lane = threadIdx.x & 31;
  const uint16_t* p = base + (size_t)(lane & 15) * ldk + ((lane >> 4) << 3);
  uint4 lo = *(const uint4*)(p);
  uint4 hi = *(const uint4*)(p + 16);
  return frag_cast(lo, hi);
}

DEV v8f wmma_bf16(v16bf a, v16bf b, v8f c) {
  return __builtin_amdgcn_wmma_f32_16x16x32_bf16(false, a, false, b,
                                                 (short)0, c, false, false);
}

// ---------------------------------------------------------------- pack kernels
__global__ __launch_bounds__(256) void cvtT_bf16_kernel(
    const float* __restrict__ src, uint16_t* __restrict__ dst,
    int logN, int K) {  // src [K][N] row-major -> dst [N][K] (transposed)
  int i = blockIdx.x * 256 + threadIdx.x;
  int k = i >> logN, n = i & ((1 << logN) - 1);
  dst[(size_t)n * K + k] = f2bf(src[i]);
}

__global__ __launch_bounds__(256) void pack_cat_kernel(
    const float* __restrict__ mems, const float* __restrict__ w,
    uint16_t* __restrict__ dst) {
  int i = blockIdx.x * 256 + threadIdx.x;            // 8192*512 elements
  float v = (i < 2097152) ? mems[i] : w[i - 2097152];
  dst[i] = f2bf(v);
}

__global__ __launch_bounds__(256) void pack_r_kernel(
    const float* __restrict__ r, uint16_t* __restrict__ dst) {
  int e = blockIdx.x * 256 + threadIdx.x;            // 1024*8*64
  int u = e >> 9, rest = e & 511, nh = rest >> 6, d = rest & 63;
  dst[((size_t)nh * 1024 + u) * 64 + d] = f2bf(r[e]); // [n][u][d]
}

// ---------------------------------------------------------------- GEMM (bf16)
// C[M,N] = A[M,K] @ BT^T, BT given N-major [N][K].  256 threads = 8 waves,
// block tile 128x128, BK=32.  A tile (128x32) streams via async LDS loads,
// double buffered; 8 WMMA per wave per k-step between barriers.
// OUT_MODE 0: f32 row-major.  1: q scatter -> [b][n][seq][d] bf16.
// 2: kv scatter -> k as [b][n][seq][d], v transposed [b][n][d][seq].
template<int OUT_MODE>
__global__ __launch_bounds__(256) void gemm_bf16_kernel(
    const uint16_t* __restrict__ A, const uint16_t* __restrict__ BT,
    int M, int N, int K, float* __restrict__ Cf,
    uint16_t* __restrict__ o1, uint16_t* __restrict__ o2) {
  // Only shared object in this kernel -> LDS offset 0 is its base.
  __shared__ __align__(16) uint16_t As[2][128 * 32];
  const int tid = threadIdx.x, lane = tid & 31, wv = tid >> 5;
  const int wm = wv & 3, wn = wv >> 2;      // 4 M-strips x 2 N-strips
  const int m0 = blockIdx.y * 128, n0 = blockIdx.x * 128;
  v8f acc[2][4] = {};

  // each thread async-copies two 16B chunks of the 128x32 A tile
  const int c0 = tid * 2;
  const int row0 = (c0) >> 2,     e0 = ((c0) & 3) << 3;
  const int row1 = (c0 + 1) >> 2, e1 = ((c0 + 1) & 3) << 3;
  const uint32_t l0 = (uint32_t)((row0 * 32 + e0) * 2);
  const uint32_t l1 = (uint32_t)((row1 * 32 + e1) * 2);

  async_load_b128(l0, A + (size_t)(m0 + row0) * K + e0);
  async_load_b128(l1, A + (size_t)(m0 + row1) * K + e1);
  wait_async0();
  __syncthreads();

  int buf = 0;
  for (int kk = 0; kk < K; kk += 32) {
    if (kk + 32 < K) {  // prefetch next A tile into the other buffer
      uint32_t nb = (uint32_t)((buf ^ 1) * 128 * 32 * 2);
      async_load_b128(nb + l0, A + (size_t)(m0 + row0) * K + kk + 32 + e0);
      async_load_b128(nb + l1, A + (size_t)(m0 + row1) * K + kk + 32 + e1);
    }
    v16bf af0 = load_a_frag(&As[buf][(wm * 32) * 32], 32);
    v16bf af1 = load_a_frag(&As[buf][(wm * 32 + 16) * 32], 32);
    #pragma unroll
    for (int ni = 0; ni < 4; ++ni) {
      const uint16_t* bp = BT + (size_t)(n0 + wn * 64 + ni * 16) * K + kk;
      if (kk + 32 < K) __builtin_prefetch(bp + 32, 0, 1);
      v16bf bf = load_bt_frag(bp, K);
      acc[0][ni] = wmma_bf16(af0, bf, acc[0][ni]);
      acc[1][ni] = wmma_bf16(af1, bf, acc[1][ni]);
    }
    wait_async0();
    __syncthreads();
    buf ^= 1;
  }

  const int lc = lane & 15, hb = (lane >> 4) << 3;
  #pragma unroll
  for (int mi = 0; mi < 2; ++mi) {
    #pragma unroll
    for (int g = 0; g < 8; ++g) {
      int R = m0 + wm * 32 + mi * 16 + g + hb;
      #pragma unroll
      for (int ni = 0; ni < 4; ++ni) {
        int C = n0 + wn * 64 + ni * 16 + lc;
        float v = acc[mi][ni][g];
        if (OUT_MODE == 0) {
          Cf[(size_t)R * N + C] = v;
        } else if (OUT_MODE == 1) {
          int seq = R >> 3, bb = R & 7, nh = C >> 6, d = C & 63;
          o1[(((size_t)(bb * 8 + nh)) * 1024 + seq) * 64 + d] = f2bf(v);
        } else {
          int seq = R >> 3, bb = R & 7;
          if (C < 512) {
            int nh = C >> 6, d = C & 63;
            o1[(((size_t)(bb * 8 + nh)) * 1024 + seq) * 64 + d] = f2bf(v);
          } else {
            int cc = C - 512, nh = cc >> 6, d = cc & 63;
            o2[(((size_t)(bb * 8 + nh)) * 64 + d) * 1024 + seq] = f2bf(v);
          }
        }
      }
    }
  }
}

// ------------------------------------------------------------- attention core
// Fused: S = (Q K^T + shift(Q R^T)) * scale, mask, softmax (scores kept in
// registers, one 16-row query tile per workgroup, 8 waves split the key dim),
// then P@V with cross-wave ds_add_f32 reduction.  Both rel-shifts reduce to
// u = j - i + 511 with the full r_emb, so one kernel serves both attentions.
template<int KLEN_T, int JPW, int MASKED>
__global__ __launch_bounds__(256) void attn_kernel(
    const uint16_t* __restrict__ q_bf, const uint16_t* __restrict__ k_bf,
    const uint16_t* __restrict__ v_bfT, const uint16_t* __restrict__ r_bf,
    int seq_off, float* __restrict__ out_f32, uint16_t* __restrict__ out_bf,
    float* __restrict__ out_denom) {
  constexpr int TILES = JPW / 16;
  constexpr int PSTEPS = JPW / 32;
  __shared__ __align__(16) float    qr_s[8][16 * 32];
  __shared__ __align__(16) uint16_t p_s[8][16 * JPW];
  __shared__ float redmax_s[8][16];
  __shared__ float redsum_s[8][16];
  __shared__ float rowsum_s[16];
  __shared__ float acc_s[16 * 64];
  const int tid = threadIdx.x, lane = tid & 31, wv = tid >> 5;
  const int i0 = blockIdx.x * 16;
  const int nh = blockIdx.y, bb = blockIdx.z;
  const int hb = (lane >> 4) << 3, lc = lane & 15;
  for (int e = tid; e < 16 * 64; e += 256) acc_s[e] = 0.0f;

  const size_t head = (size_t)(bb * 8 + nh) * 1024;
  const uint16_t* qp = q_bf + (head + seq_off + i0) * 64;
  v16bf a0 = load_a_frag(qp, 64);        // Q dims 0..31
  v16bf a1 = load_a_frag(qp + 32, 64);   // Q dims 32..63

  v8f acs[TILES];
  const uint16_t* kbase = k_bf + head * 64;
  const uint16_t* rbase = r_bf + (size_t)nh * 1024 * 64;
  #pragma unroll
  for (int jt = 0; jt < TILES; ++jt) {
    int j0 = wv * JPW + jt * 16;
    v8f s = {};
    const uint16_t* kb = kbase + (size_t)j0 * 64;
    s = wmma_bf16(a0, load_bt_frag(kb, 64), s);
    s = wmma_bf16(a1, load_bt_frag(kb + 32, 64), s);
    // Q R^T on u-window [ubase, ubase+32) -- 16-aligned since i0,j0 are.
    int ubase = j0 - i0 + 496;
    #pragma unroll
    for (int ut = 0; ut < 2; ++ut) {
      int u0 = ubase + ut * 16;
      if (u0 >= 0 && u0 + 16 <= 1024) {   // OOB tiles feed only masked elems
        v8f qr = {};
        const uint16_t* rb = rbase + (size_t)u0 * 64;
        qr = wmma_bf16(a0, load_bt_frag(rb, 64), qr);
        qr = wmma_bf16(a1, load_bt_frag(rb + 32, 64), qr);
        #pragma unroll
        for (int g = 0; g < 8; ++g)
          qr_s[wv][(g + hb) * 32 + ut * 16 + lc] = qr[g];
      }
    }
    #pragma unroll
    for (int g = 0; g < 8; ++g) {
      int lr = g + hb, i = i0 + lr, j = j0 + lc;
      float qr = qr_s[wv][lr * 32 + (lc - lr + 15)];  // u - ubase in [0,30]
      float sv = (s[g] + qr) * 0.125f;                // 1/sqrt(64)
      bool masked = MASKED ? (j > i + 512) : (j >= KLEN_T);
      acs[jt][g] = masked ? -3.0e38f : sv;
    }
  }

  // ---- softmax: row stats via shfl (16-lane halves) + LDS across waves
  float gm[8], ps[8];
  #pragma unroll
  for (int g = 0; g < 8; ++g) {
    float m = -3.0e38f;
    #pragma unroll
    for (int jt = 0; jt < TILES; ++jt) m = fmaxf(m, acs[jt][g]);
    #pragma unroll
    for (int d = 1; d < 16; d <<= 1) m = fmaxf(m, __shfl_xor(m, d, 32));
    gm[g] = m;
  }
  if ((lane & 15) == 0) {
    #pragma unroll
    for (int g = 0; g < 8; ++g) redmax_s[wv][hb + g] = gm[g];
  }
  __syncthreads();
  #pragma unroll
  for (int g = 0; g < 8; ++g) {
    float m = redmax_s[0][hb + g];
    #pragma unroll
    for (int w2 = 1; w2 < 8; ++w2) m = fmaxf(m, redmax_s[w2][hb + g]);
    gm[g] = m;
    ps[g] = 0.0f;
  }
  #pragma unroll
  for (int jt = 0; jt < TILES; ++jt) {
    #pragma unroll
    for (int g = 0; g < 8; ++g) {
      float e = __expf(acs[jt][g] - gm[g]);
      ps[g] += e;
      p_s[wv][(g + hb) * JPW + jt * 16 + lc] = f2bf(e);
    }
  }
  #pragma unroll
  for (int g = 0; g < 8; ++g) {
    #pragma unroll
    for (int d = 1; d < 16; d <<= 1) ps[g] += __shfl_xor(ps[g], d, 32);
  }
  if ((lane & 15) == 0) {
    #pragma unroll
    for (int g = 0; g < 8; ++g) redsum_s[wv][hb + g] = ps[g];
  }
  __syncthreads();
  #pragma unroll
  for (int g = 0; g < 8; ++g) {
    float ssum = 0.0f;
    #pragma unroll
    for (int w2 = 0; w2 < 8; ++w2) ssum += redsum_s[w2][hb + g];
    ps[g] = ssum;
  }
  if (wv == 0 && (lane & 15) == 0) {
    #pragma unroll
    for (int g = 0; g < 8; ++g) {
      rowsum_s[hb + g] = ps[g];
      int i = i0 + hb + g;
      out_denom[((size_t)i * 8 + bb) * 8 + nh] = gm[g] + __logf(ps[g]);
    }
  }

  // ---- P @ V (unnormalized exp; scale by 1/rowsum at the end)
  const uint16_t* vb = v_bfT + (size_t)(bb * 8 + nh) * 64 * 1024;
  v8f pv[4] = {};
  #pragma unroll
  for (int ks = 0; ks < PSTEPS; ++ks) {
    v16bf pa = load_a_frag(&p_s[wv][ks * 32], JPW);
    #pragma unroll
    for (int t = 0; t < 4; ++t) {
      const uint16_t* vp = vb + (size_t)(t * 16) * 1024 + wv * JPW + ks * 32;
      pv[t] = wmma_bf16(pa, load_bt_frag(vp, 1024), pv[t]);
    }
  }
  #pragma unroll
  for (int t = 0; t < 4; ++t) {
    #pragma unroll
    for (int g = 0; g < 8; ++g)
      atomicAdd(&acc_s[(g + hb) * 64 + t * 16 + lc], pv[t][g]);
  }
  __syncthreads();
  for (int e = tid; e < 16 * 64; e += 256) {
    int lr = e >> 6, d = e & 63, i = i0 + lr;
    float v = acc_s[e] / rowsum_s[lr];
    out_f32[(((size_t)i * 8 + bb) * 8 + nh) * 64 + d] = v;
    if (MASKED)  // main attention also feeds the Wo GEMM in bf16
      out_bf[((size_t)i * 8 + bb) * 512 + nh * 64 + d] = f2bf(v);
  }
}

// ---------------------------------------------------------------- LaMemo gate
__global__ __launch_bounds__(256) void alpha_kernel(
    const float* __restrict__ denom, const float* __restrict__ memden,
    const float* __restrict__ recur, const float* __restrict__ avm,
    uint16_t* __restrict__ im_bf) {
  int idx = blockIdx.x * 256 + threadIdx.x;          // 512*8*8
  if (idx >= 32768) return;
  float dn = denom[idx], md = memden[idx];
  float mx = fmaxf(dn, md), mn = fminf(dn, md);
  float lae = mx + log1pf(__expf(mn - mx));          // logaddexp
  float alpha = __expf(dn - lae);
  int i = idx >> 6, bb = (idx >> 3) & 7, nh = idx & 7;
  size_t rb = (size_t)idx * 64;
  size_t ob = ((size_t)(i * 8 + bb)) * 512 + nh * 64;
  for (int d = 0; d < 64; ++d) {
    float inter = recur[rb + d] * alpha + avm[rb + d] * (1.0f - alpha);
    im_bf[ob + d] = f2bf(inter);
  }
}

// ------------------------------------------------------------------ layernorm
__global__ __launch_bounds__(256) void ln_kernel(
    const float* __restrict__ res, const float* __restrict__ proj,
    const float* __restrict__ g, const float* __restrict__ b,
    float* __restrict__ out) {
  __shared__ float red[8];
  const int tid = threadIdx.x;
  const size_t base = (size_t)blockIdx.x * 512;
  float x0 = res[base + tid * 2]     + proj[base + tid * 2];
  float x1 = res[base + tid * 2 + 1] + proj[base + tid * 2 + 1];
  float s = x0 + x1;
  #pragma unroll
  for (int d = 1; d < 32; d <<= 1) s += __shfl_xor(s, d, 32);
  if ((tid & 31) == 0) red[tid >> 5] = s;
  __syncthreads();
  float tot = 0.0f;
  #pragma unroll
  for (int w2 = 0; w2 < 8; ++w2) tot += red[w2];
  float mean = tot * (1.0f / 512.0f);
  float d0 = x0 - mean, d1 = x1 - mean;
  float vs = d0 * d0 + d1 * d1;
  __syncthreads();
  #pragma unroll
  for (int d = 1; d < 32; d <<= 1) vs += __shfl_xor(vs, d, 32);
  if ((tid & 31) == 0) red[tid >> 5] = vs;
  __syncthreads();
  float vtot = 0.0f;
  #pragma unroll
  for (int w2 = 0; w2 < 8; ++w2) vtot += red[w2];
  float rstd = rsqrtf(vtot * (1.0f / 512.0f) + 1e-5f);
  out[base + tid * 2]     = d0 * rstd * g[tid * 2]     + b[tid * 2];
  out[base + tid * 2 + 1] = d1 * rstd * g[tid * 2 + 1] + b[tid * 2 + 1];
}

// ---------------------------------------------------------------------- host
extern "C" void kernel_launch(void* const* d_in, const int* in_sizes, int n_in,
                              void* d_out, int out_size, void* d_ws, size_t ws_size,
                              hipStream_t stream) {
  (void)in_sizes; (void)n_in; (void)out_size; (void)ws_size;
  const float* w     = (const float*)d_in[0];
  const float* r_emb = (const float*)d_in[1];
  // d_in[2] attn_mask: recomputed analytically (j > i + MLEN)
  const float* mems  = (const float*)d_in[3];
  const float* recur = (const float*)d_in[4];
  const float* denom = (const float*)d_in[5];
  const float* Wq    = (const float*)d_in[6];
  const float* Wkv   = (const float*)d_in[7];
  const float* Wo    = (const float*)d_in[8];
  const float* ln_g  = (const float*)d_in[9];
  const float* ln_b  = (const float*)d_in[10];

  float* out       = (float*)d_out;
  float* out_mem   = out + 2097152;
  float* out_recur = out + 4194304;
  float* out_denom = out + 6291456;

  uint8_t* W = (uint8_t*)d_ws;
  uint16_t* cat_bf = (uint16_t*)(W + 0);          //  8 MB
  uint16_t* wqT    = (uint16_t*)(W + 8388608);    //  [512][512]  N-major
  uint16_t* wkvT   = (uint16_t*)(W + 8912896);    //  [1024][512] N-major
  uint16_t* woT    = (uint16_t*)(W + 9961472);    //  [512][512]  N-major
  uint16_t* r_bf   = (uint16_t*)(W + 10485760);
  uint16_t* q_bf   = (uint16_t*)(W + 11534336);   //  8 MB [b][n][seq][d]
  uint16_t* k_bf   = (uint16_t*)(W + 19922944);   //  8 MB [b][n][seq][d]
  uint16_t* v_bfT  = (uint16_t*)(W + 28311552);   //  8 MB [b][n][d][seq]
  uint16_t* av_bf  = (uint16_t*)(W + 36700160);   //  4 MB
  uint16_t* im_bf  = (uint16_t*)(W + 40894464);   //  4 MB
  float*    avm    = (float*)(W + 45088768);      //  8 MB
  float*    memden = (float*)(W + 53477376);      // 128 KB
  float*    ao     = (float*)(W + 0);             // reuses cat_bf (dead)
  float*    aom    = (float*)(W + 19922944);      // reuses k_bf (dead)

  // pack to bf16 (weights transposed to N-major for direct B-frag loads)
  pack_cat_kernel<<<16384, 256, 0, stream>>>(mems, w, cat_bf);
  cvtT_bf16_kernel<<<1024, 256, 0, stream>>>(Wq, wqT, 9, 512);
  cvtT_bf16_kernel<<<2048, 256, 0, stream>>>(Wkv, wkvT, 10, 512);
  cvtT_bf16_kernel<<<1024, 256, 0, stream>>>(Wo, woT, 9, 512);
  pack_r_kernel<<<2048, 256, 0, stream>>>(r_emb, r_bf);

  // projections: q_all / (k_all, v_all)
  gemm_bf16_kernel<1><<<dim3(4, 64), 256, 0, stream>>>(
      cat_bf, wqT, 8192, 512, 512, nullptr, q_bf, nullptr);
  gemm_bf16_kernel<2><<<dim3(8, 64), 256, 0, stream>>>(
      cat_bf, wkvT, 8192, 1024, 512, nullptr, k_bf, v_bfT);

  // attentions: main (masked, klen=1024) and memory (klen=513)
  dim3 ga(32, 8, 8);
  attn_kernel<1024, 128, 1><<<ga, 256, 0, stream>>>(
      q_bf, k_bf, v_bfT, r_bf, 512, out_recur, av_bf, out_denom);
  attn_kernel<513, 96, 0><<<ga, 256, 0, stream>>>(
      q_bf, k_bf, v_bfT, r_bf, 0, avm, nullptr, memden);

  // LaMemo interpolation -> inter_mems (bf16 for Wo GEMM)
  alpha_kernel<<<128, 256, 0, stream>>>(denom, memden, recur, avm, im_bf);

  // output projections
  gemm_bf16_kernel<0><<<dim3(4, 32), 256, 0, stream>>>(
      av_bf, woT, 4096, 512, 512, ao, nullptr, nullptr);
  gemm_bf16_kernel<0><<<dim3(4, 32), 256, 0, stream>>>(
      im_bf, woT, 4096, 512, 512, aom, nullptr, nullptr);

  // residual + layernorm
  ln_kernel<<<4096, 256, 0, stream>>>(w, ao, ln_g, ln_b, out);
  ln_kernel<<<4096, 256, 0, stream>>>(mems, aom, ln_g, ln_b, out_mem);
}
